// GraphConvolutionalNetwork_8409545965691
// MI455X (gfx1250) — matrix operand order repaired and checked by reading, not compiled, not run on hardware
//
#include <hip/hip_runtime.h>

// Problem constants (match reference)
static constexpr int BB  = 8;     // batch
static constexpr int SS  = 512;   // sequence (feature dim per node)
static constexpr int NN  = 2048;  // nodes
static constexpr int CH  = 512;   // HID == SS == 512, channels of both GEMMs

typedef _Float16 v16h __attribute__((ext_vector_type(16)));
typedef float    v8f  __attribute__((ext_vector_type(8)));
typedef _Float16 h4   __attribute__((ext_vector_type(4)));

union Frag {
    v16h v;
    uint4 q[2];
    _Float16 h[16];
};

// CDNA5 async global->LDS copy (ASYNCcnt-tracked), 16B per lane.
__device__ __forceinline__ void async_copy_b128(uint32_t lds_byte_off, const void* gptr) {
    asm volatile("global_load_async_to_lds_b128 %0, %1, off"
                 :: "v"(lds_byte_off), "v"(gptr) : "memory");
}
__device__ __forceinline__ void wait_async0() {
    asm volatile("s_wait_asynccnt 0x0" ::: "memory");
}

// ---------------------------------------------------------------------------
// 1) fv[b][n] = sum_s x[b,s,n]  (positive scalings preserve sign & ordering,
//    which is all the adjacency construction depends on)
// ---------------------------------------------------------------------------
__global__ __launch_bounds__(256) void fv_kernel(const float* __restrict__ x,
                                                 float* __restrict__ fv) {
    int b = blockIdx.y;
    int n = blockIdx.x * 256 + threadIdx.x;
    const float* xb = x + (size_t)b * SS * NN + n;
    float s = 0.f;
    for (int j = 0; j < SS; ++j) s += xb[(size_t)j * NN];
    fv[b * NN + n] = s;
}

// ---------------------------------------------------------------------------
// 2) Per-batch stable top-9 / bottom-9 of fv (jax tie-break: lower index
//    first) + pos/neg counts.  meta[b*32+0..8]=top, +9..17=bottom, 18=#pos,
//    19=#nonpos.
// ---------------------------------------------------------------------------
__global__ void topk_kernel(const float* __restrict__ fv, int* __restrict__ meta) {
    int b = threadIdx.x;
    if (b >= BB) return;
    const float* f = fv + b * NN;
    float tv[9], bv[9];
    int   ti[9], bi[9];
    for (int r = 0; r < 9; ++r) {
        tv[r] = -__builtin_huge_valf(); ti[r] = 0x7fffffff;
        bv[r] =  __builtin_huge_valf(); bi[r] = 0x7fffffff;
    }
    int np = 0, nn = 0;
    for (int j = 0; j < NN; ++j) {
        float v = f[j];
        if (v > 0.f) ++np; else ++nn;
        for (int r = 0; r < 9; ++r) {
            if (v > tv[r] || (v == tv[r] && j < ti[r])) {
                for (int s = 8; s > r; --s) { tv[s] = tv[s-1]; ti[s] = ti[s-1]; }
                tv[r] = v; ti[r] = j; break;
            }
        }
        for (int r = 0; r < 9; ++r) {
            if (v < bv[r] || (v == bv[r] && j < bi[r])) {
                for (int s = 8; s > r; --s) { bv[s] = bv[s-1]; bi[s] = bi[s-1]; }
                bv[r] = v; bi[r] = j; break;
            }
        }
    }
    int* mb = meta + b * 32;
    for (int r = 0; r < 9; ++r) { mb[r] = ti[r]; mb[9 + r] = bi[r]; }
    mb[18] = np; mb[19] = nn;
}

// ---------------------------------------------------------------------------
// 2b) f32 -> f16 weight conversion (vectorized, n multiple of 1024)
// ---------------------------------------------------------------------------
__global__ __launch_bounds__(256) void tohalf4_kernel(const float* __restrict__ src,
                                                      _Float16* __restrict__ dst) {
    int i = (blockIdx.x * 256 + threadIdx.x) * 4;
    float4 v = *(const float4*)(src + i);
    h4 hv = { (_Float16)v.x, (_Float16)v.y, (_Float16)v.z, (_Float16)v.w };
    *(h4*)(dst + i) = hv;
}

// ---------------------------------------------------------------------------
// 3) Tiled f16 WMMA GEMM:  C[m,n] = sum_k A(m,k) * W[n*K + k] + bias[n]
//    AMODE==0: A(m,k) = A32[k*Mrows + m]  (f32 col-major; convert through LDS)
//    AMODE==1: A(m,k) = A16[m*K + k]      (f16 row-major; async copy to LDS)
//    B (f16) is always staged with GLOBAL_LOAD_ASYNC_TO_LDS_B128.
//    Tile: 128(M) x 128(N) x 32(K); 8 waves; each wave 64x32 (4x2 WMMA accs).
// ---------------------------------------------------------------------------
template <int AMODE>
__global__ __launch_bounds__(256) void gemm_kernel(const float* __restrict__ A32,
                                                   const _Float16* __restrict__ A16,
                                                   const _Float16* __restrict__ W,
                                                   const float* __restrict__ bias,
                                                   float* __restrict__ C,
                                                   int Mrows, int K, int Nout,
                                                   long strideA, long strideC) {
    constexpr int LDSP = 40;  // padded half stride (32+8) -> 80B rows, 16B aligned
    __shared__ __align__(16) _Float16 As[128 * LDSP];
    __shared__ __align__(16) _Float16 Bs[128 * LDSP];

    const int b = blockIdx.z;
    const float*    A32b = (AMODE == 0) ? (A32 + (size_t)b * strideA) : nullptr;
    const _Float16* A16b = (AMODE == 1) ? (A16 + (size_t)b * strideA) : nullptr;
    float* Cb = C + (size_t)b * strideC;

    const uint32_t AsOff = (uint32_t)(uintptr_t)&As[0];
    const uint32_t BsOff = (uint32_t)(uintptr_t)&Bs[0];

    const int m0 = blockIdx.x * 128;
    const int n0 = blockIdx.y * 128;
    const int tid  = threadIdx.x;
    const int lane = tid & 31;
    const int wave = tid >> 5;
    const int wm = wave & 1;   // M position of wave (0..1)
    const int wn = wave >> 1;  // N position of wave (0..3)
    const int lm = lane & 15;
    const int kh = lane >> 4;  // half-wave K selector (CDNA5 16-bit layout)

    v8f acc[4][2];
    for (int mi = 0; mi < 4; ++mi)
        for (int ni = 0; ni < 2; ++ni)
            acc[mi][ni] = v8f{0.f, 0.f, 0.f, 0.f, 0.f, 0.f, 0.f, 0.f};

    for (int k0 = 0; k0 < K; k0 += 32) {
        // ---- stage A tile (128 rows x 32 halves, row-major, stride 80B) ----
        if constexpr (AMODE == 0) {
            for (int it = 0; it < 4; ++it) {
                int idx = tid + it * 256;          // 1024 float4s
                int kk  = idx >> 5;                // 0..31
                int mc  = (idx & 31) << 2;         // 0..124
                float4 v = *(const float4*)(A32b + (size_t)(k0 + kk) * Mrows + m0 + mc);
                As[(mc + 0) * LDSP + kk] = (_Float16)v.x;
                As[(mc + 1) * LDSP + kk] = (_Float16)v.y;
                As[(mc + 2) * LDSP + kk] = (_Float16)v.z;
                As[(mc + 3) * LDSP + kk] = (_Float16)v.w;
            }
        } else {
            for (int it = 0; it < 2; ++it) {
                int id  = tid + it * 256;          // 512 x 16B chunks
                int row = id >> 2;                 // 0..127
                int kc  = id & 3;                  // 0..3 (8 halves each)
                async_copy_b128(AsOff + row * 80 + kc * 16,
                                A16b + (size_t)(m0 + row) * K + k0 + kc * 8);
            }
        }
        // ---- stage B tile: col-major per output column, 32 contiguous K ----
        for (int it = 0; it < 2; ++it) {
            int id  = tid + it * 256;
            int col = id >> 2;                     // 0..127
            int kc  = id & 3;
            async_copy_b128(BsOff + col * 80 + kc * 16,
                            W + (size_t)(n0 + col) * K + k0 + kc * 8);
        }
        wait_async0();
        __syncthreads();

        // ---- B frags: lane holds column lm; K {0-7,16-23} / {8-15,24-31}
        Frag fb[2];
        for (int ni = 0; ni < 2; ++ni) {
            const _Float16* bp = &Bs[(wn * 32 + ni * 16 + lm) * LDSP + kh * 8];
            fb[ni].q[0] = *(const uint4*)(bp);
            fb[ni].q[1] = *(const uint4*)(bp + 16);
        }
        // ---- A frags: lane holds row lm; K {0-7,16-23} / {8-15,24-31}
        for (int mi = 0; mi < 4; ++mi) {
            const _Float16* ap = &As[(wm * 64 + mi * 16 + lm) * LDSP + kh * 8];
            Frag fa;
            fa.q[0] = *(const uint4*)(ap);
            fa.q[1] = *(const uint4*)(ap + 16);
            for (int ni = 0; ni < 2; ++ni) {
                acc[mi][ni] = __builtin_amdgcn_wmma_f32_16x16x32_f16(
                    false, fa.v, false, fb[ni].v, (short)0, acc[mi][ni], false, false);
            }
        }
        __syncthreads();
    }

    // ---- epilogue: bias + store (C/D: lane<16 -> M=r, lane>=16 -> M=8+r) ----
    for (int ni = 0; ni < 2; ++ni) {
        int colg = n0 + wn * 32 + ni * 16 + lm;
        float bvv = bias[colg];
        for (int mi = 0; mi < 4; ++mi) {
            for (int r = 0; r < 8; ++r) {
                int rowg = m0 + wm * 64 + mi * 16 + (kh ? 8 + r : r);
                Cb[(size_t)rowg * Nout + colg] = acc[mi][ni][r] + bvv;
            }
        }
    }
}

// ---------------------------------------------------------------------------
// 4) Per-batch channel sums over structured vertex classes:
//    [0]=sum{fv>0}, [1]=sum{fv<=0}, [2]=P-set(ranks 1..8), [3]=N-set
// ---------------------------------------------------------------------------
__global__ __launch_bounds__(128) void sums_kernel(const float* __restrict__ H,
                                                   const float* __restrict__ fv,
                                                   const int* __restrict__ meta,
                                                   float* __restrict__ sums) {
    int b = blockIdx.y;
    int c = blockIdx.x * 128 + threadIdx.x;
    const float* Hb = H + (size_t)b * NN * CH;
    const float* f  = fv + b * NN;
    float sp = 0.f, sn = 0.f;
    for (int j = 0; j < NN; ++j) {
        float v = Hb[(size_t)j * CH + c];
        if (f[j] > 0.f) sp += v; else sn += v;
    }
    const int* mb = meta + b * 32;
    float sP = 0.f, sN = 0.f;
    for (int r = 1; r < 9; ++r) {
        sP += Hb[(size_t)mb[r]     * CH + c];
        sN += Hb[(size_t)mb[9 + r] * CH + c];
    }
    float* sb = sums + (size_t)b * 4 * CH;
    sb[0 * CH + c] = sp;
    sb[1 * CH + c] = sn;
    sb[2 * CH + c] = sP;
    sb[3 * CH + c] = sN;
}

// ---------------------------------------------------------------------------
// Structured degree-normalized aggregation for element (i, c).
// ---------------------------------------------------------------------------
__device__ __forceinline__ float agg_value(int i, int c,
                                           const float* __restrict__ Hb,
                                           const float* __restrict__ f,
                                           const int* __restrict__ mb,
                                           const float* __restrict__ sb) {
    float fi = f[i];
    bool pos = fi > 0.f;
    int hub = 0;  // 0 = normal, 1 = in P-set, 2 = in N-set
    for (int r = 1; r < 9; ++r) if (mb[r]     == i) hub = 1;
    for (int r = 1; r < 9; ++r) if (mb[9 + r] == i) hub = 2;

    if (hub == 0) {
        float s = Hb[(size_t)i * CH + c] + (pos ? sb[2 * CH + c] : sb[3 * CH + c]);
        return s * (1.f / 9.f);
    }
    const int* Sset = pos ? (mb + 1) : (mb + 10);
    float sum;
    int   deg;
    if (hub == 1) { sum = sb[0 * CH + c]; deg = mb[18]; }
    else          { sum = sb[1 * CH + c]; deg = mb[19]; }
    for (int r = -1; r < 8; ++r) {
        int j = (r < 0) ? i : Sset[r];
        if (r >= 0 && j == i) continue;                 // dedupe {i} vs S_i
        bool jpos = f[j] > 0.f;
        bool alreadyCounted = (hub == 1) ? jpos : !jpos;
        if (!alreadyCounted) {
            sum += Hb[(size_t)j * CH + c];
            deg += 1;
        }
    }
    return sum / (float)deg;
}

// Layer 1: aggregate + ReLU, stores f16 row-major (N, CH) for the f16 GEMM2.
__global__ __launch_bounds__(256) void agg1_kernel(const float* __restrict__ H,
                                                   const float* __restrict__ fv,
                                                   const int* __restrict__ meta,
                                                   const float* __restrict__ sums,
                                                   _Float16* __restrict__ Out) {
    int b  = blockIdx.y;
    int id = blockIdx.x * 256 + threadIdx.x;
    int c  = id & (CH - 1);
    int i  = id >> 9;
    const float* Hb = H + (size_t)b * NN * CH;
    const float* f  = fv + b * NN;
    const int*   mb = meta + b * 32;
    const float* sb = sums + (size_t)b * 4 * CH;
    float v = agg_value(i, c, Hb, f, mb, sb);
    Out[(size_t)b * NN * CH + (size_t)i * CH + c] = (_Float16)fmaxf(v, 0.f);
}

// Layer 2: aggregate + transposed store to out (B, S, N) via LDS 32x32 tile.
__global__ __launch_bounds__(256) void agg2_kernel(const float* __restrict__ H,
                                                   const float* __restrict__ fv,
                                                   const int* __restrict__ meta,
                                                   const float* __restrict__ sums,
                                                   float* __restrict__ out) {
    __shared__ float t[32][33];
    int b  = blockIdx.z;
    int i0 = blockIdx.x * 32;
    int c0 = blockIdx.y * 32;
    const float* Hb = H + (size_t)b * NN * CH;
    const float* f  = fv + b * NN;
    const int*   mb = meta + b * 32;
    const float* sb = sums + (size_t)b * 4 * CH;

    int lc = threadIdx.x & 31;
    int r0 = threadIdx.x >> 5;
    for (int rr = 0; rr < 4; ++rr) {
        int ir = r0 + rr * 8;
        t[ir][lc] = agg_value(i0 + ir, c0 + lc, Hb, f, mb, sb);
    }
    __syncthreads();
    int li = threadIdx.x & 31;
    for (int rr = 0; rr < 4; ++rr) {
        int cr = r0 + rr * 8;
        out[(size_t)b * SS * NN + (size_t)(c0 + cr) * NN + i0 + li] = t[cr][li];
    }
}

// ---------------------------------------------------------------------------
extern "C" void kernel_launch(void* const* d_in, const int* in_sizes, int n_in,
                              void* d_out, int out_size, void* d_ws, size_t ws_size,
                              hipStream_t stream) {
    (void)in_sizes; (void)n_in; (void)out_size; (void)ws_size;
    const float* x  = (const float*)d_in[0];   // (B, S, N)
    const float* W1 = (const float*)d_in[1];   // (HID, S)
    const float* b1 = (const float*)d_in[2];   // (HID)
    const float* W2 = (const float*)d_in[3];   // (S, HID)
    const float* b2 = (const float*)d_in[4];   // (S)
    float* out = (float*)d_out;                // (B, S, N)

    char* ws = (char*)d_ws;
    float*     fv   = (float*)    (ws);                           // 64 KB
    int*       meta = (int*)      (ws + (64 << 10));              //  1 KB
    float*     sums = (float*)    (ws + (68 << 10));              // 64 KB
    _Float16*  Wh1  = (_Float16*) (ws + (256 << 10));             // 512 KB
    _Float16*  Wh2  = (_Float16*) (ws + (768 << 10));             // 512 KB
    float*     bufA = (float*)    (ws + ((size_t)2 << 20));       // 32 MB (lin out, f32)
    _Float16*  bufB = (_Float16*) (ws + ((size_t)34 << 20));      // 16 MB (H1, f16)

    // Graph structure
    fv_kernel<<<dim3(NN / 256, BB), 256, 0, stream>>>(x, fv);
    topk_kernel<<<1, 32, 0, stream>>>(fv, meta);

    // Weights -> f16 once
    tohalf4_kernel<<<(CH * SS) / 1024, 256, 0, stream>>>(W1, Wh1);
    tohalf4_kernel<<<(CH * SS) / 1024, 256, 0, stream>>>(W2, Wh2);

    // Layer 1: WMMA GEMM -> class sums -> aggregate + ReLU (f16 out)
    gemm_kernel<0><<<dim3(NN / 128, CH / 128, BB), 256, 0, stream>>>(
        x, nullptr, Wh1, b1, bufA, NN, SS, CH, (long)SS * NN, (long)NN * CH);
    sums_kernel<<<dim3(CH / 128, BB), 128, 0, stream>>>(bufA, fv, meta, sums);
    agg1_kernel<<<dim3((NN * CH) / 256, BB), 256, 0, stream>>>(bufA, fv, meta, sums, bufB);

    // Layer 2: WMMA GEMM (f16 A via async) -> sums -> aggregate + transpose
    gemm_kernel<1><<<dim3(NN / 128, CH / 128, BB), 256, 0, stream>>>(
        nullptr, bufB, Wh2, b2, bufA, NN, CH, SS, (long)NN * CH, (long)NN * CH);
    sums_kernel<<<dim3(CH / 128, BB), 128, 0, stream>>>(bufA, fv, meta, sums);
    agg2_kernel<<<dim3(NN / 32, CH / 32, BB), 256, 0, stream>>>(bufA, fv, meta, sums, out);
}
